// EquivariantGraphConvolution_50792283242913
// MI455X (gfx1250) — matrix operand order repaired and checked by reading, not compiled, not run on hardware
//
#include <hip/hip_runtime.h>

// ---------------------------------------------------------------------------
// EGNN layer for MI455X (gfx1250, wave32, WMMA f32_16x16x32_f16).
// Memory-bound workload (~900MB traffic @ 23.3TB/s ≈ 40us floor); matmuls are
// done with WMMA on f16 inputs / f32 accumulate. Weights are pre-packed into
// per-lane B-fragment order so each B tile is one coalesced 32B load.
// ---------------------------------------------------------------------------

typedef __attribute__((ext_vector_type(16))) _Float16 v16h;
typedef __attribute__((ext_vector_type(8)))  float    v8f;
typedef _Float16 half_t;

#define NN    50000
#define NE    1000000
#define ROW   70      // node row: 3 coords + 3 vels + 64 nf
#define NFOFF 6

__device__ __forceinline__ float silu_f(float x){ return x / (1.0f + __expf(-x)); }
__device__ __forceinline__ float sigm_f(float x){ return 1.0f / (1.0f + __expf(-x)); }

__device__ __forceinline__ v8f wmma16(v16h a, v16h b, v8f c){
  // (neg_a, A, neg_b, B, c_mod, C, reuse_a, reuse_b)
  return __builtin_amdgcn_wmma_f32_16x16x32_f16(false, a, false, b, (short)0, c,
                                                false, false);
}

// B fragment: pre-packed [kt][nt][lane][16 halfs]; one 32B vector load.
__device__ __forceinline__ v16h bfrag(const half_t* __restrict__ w, int lane,
                                      int kt, int nt){
  return *(const v16h*)(w + (size_t)(((kt * 4 + nt) * 32 + lane) * 16));
}

// A fragment from a 16x64 f16 LDS tile, per ISA 16-bit A layout:
// lane m=lane&15; halves 0-7 -> K = kb+0..7, halves 8-15 -> K = kb+16..23,
// kb = kt*32 + (lane>=16 ? 8 : 0).
__device__ __forceinline__ v16h afrag_lds(const half_t* buf, int lane, int kt){
  const int m  = lane & 15;
  const int kb = kt * 32 + ((lane & 16) ? 8 : 0);
  v16h a;
#pragma unroll
  for (int i = 0; i < 16; ++i){
    int k = kb + (i & 7) + ((i & 8) ? 16 : 0);
    a[i] = buf[m * 64 + k];
  }
  return a;
}

// Apply SiLU to 4 N-tile accumulators (f32 C/D layout), keep f32 result in
// place, and store f16 copy to a 16x64 LDS tile for A-relayout.
__device__ __forceinline__ void silu_store_lds(v8f* acc, half_t* buf, int lane){
  const int hi  = (lane >> 4) & 1;
  const int n15 = lane & 15;
#pragma unroll
  for (int nt = 0; nt < 4; ++nt){
#pragma unroll
    for (int r = 0; r < 8; ++r){
      float v = silu_f(acc[nt][r]);
      acc[nt][r] = v;
      buf[(r + hi * 8) * 64 + nt * 16 + n15] = (half_t)v;
    }
  }
}

// Reduce across the 16 lanes of a half-wave (N dimension of C/D layout).
__device__ __forceinline__ float red16(float v){
  v += __shfl_xor(v, 1, 32);
  v += __shfl_xor(v, 2, 32);
  v += __shfl_xor(v, 4, 32);
  v += __shfl_xor(v, 8, 32);
  return v;
}

// ---------------------------------------------------------------------------
// Pack a row-major [K x 64] f32 weight into f16 WMMA B-fragment order.
// One thread per (kt, nt, lane) -> writes 16 halves.
// ---------------------------------------------------------------------------
__global__ void pack_bfrag_kernel(const float* __restrict__ W,
                                  half_t* __restrict__ dst, int K, int kts){
  int tid = blockIdx.x * blockDim.x + threadIdx.x;
  int total = kts * 128;
  if (tid >= total) return;
  int kt   = tid >> 7;
  int rem  = tid & 127;
  int lane = rem & 31;
  int n    = (rem >> 5) * 16 + (lane & 15);
  int kb   = kt * 32 + ((lane & 16) ? 8 : 0);
  half_t* p = dst + (size_t)tid * 16;
#pragma unroll
  for (int i = 0; i < 16; ++i){
    int k = kb + (i & 7) + ((i & 8) ? 16 : 0);
    float v = (k < K) ? W[(size_t)k * 64 + n] : 0.0f;
    p[i] = (half_t)v;
  }
}

__global__ void zero_f32_kernel(float* __restrict__ p, int n){
  int i = blockIdx.x * blockDim.x + threadIdx.x;
  if (i < n) p[i] = 0.0f;
}

// ---------------------------------------------------------------------------
// Edge kernel: 1 wave = 16 edges. GEMM chain + reductions + atomic scatter.
// ---------------------------------------------------------------------------
__global__ __launch_bounds__(256) void egnn_edge_kernel(
    const float* __restrict__ nfi, const int* __restrict__ eidx,
    const float* __restrict__ ef,
    const half_t* __restrict__ We1f, const float* __restrict__ be1,
    const half_t* __restrict__ We2f, const float* __restrict__ be2,
    const half_t* __restrict__ Wc1f, const float* __restrict__ bc1,
    const float* __restrict__ Wc2, const float* __restrict__ Wi,
    const float* __restrict__ bi,
    float* __restrict__ agg, float* __restrict__ csum, float* __restrict__ cnt)
{
  __shared__ half_t ldsT[8][16 * 64];   // per-wave 16x64 f16 relayout tile
  __shared__ float  ldsD[8][16 * 4];    // per-wave {dx,dy,dz,norm} per edge

  const int wid  = threadIdx.x >> 5;
  const int lane = threadIdx.x & 31;
  const int e0   = (blockIdx.x * 8 + wid) * 16;
  if (e0 >= NE) return;

  const int* startv = eidx;
  const int* endv   = eidx + NE;
  const int m   = lane & 15;
  const int hi  = (lane >> 4) & 1;
  const int n15 = lane & 15;
  const int em  = e0 + m;
  const int sm  = startv[em];
  const int tm  = endv[em];

  if (lane < 16){
    const float* cs = nfi + (size_t)sm * ROW;
    const float* ct = nfi + (size_t)tm * ROW;
    float dx = cs[0] - ct[0], dy = cs[1] - ct[1], dz = cs[2] - ct[2];
    ldsD[wid][m * 4 + 0] = dx;
    ldsD[wid][m * 4 + 1] = dy;
    ldsD[wid][m * 4 + 2] = dz;
    ldsD[wid][m * 4 + 3] = sqrtf(dx * dx + dy * dy + dz * dz);
  }
  asm volatile("s_wait_dscnt 0" ::: "memory");
  const float nrm = ldsD[wid][m * 4 + 3];

  const float* srow = nfi + (size_t)sm * ROW + NFOFF;
  const float* trow = nfi + (size_t)tm * ROW + NFOFF;
  const float* erow = ef  + (size_t)em * 16;

  // ---- GEMM1: h = silu(x @ W_e1 + b_e1), K=145 padded to 160 (5 k-tiles)
  v8f acc1[4];
#pragma unroll
  for (int nt = 0; nt < 4; ++nt){
    float b = be1[nt * 16 + n15];
#pragma unroll
    for (int r = 0; r < 8; ++r) acc1[nt][r] = b;
  }
#pragma unroll
  for (int kt = 0; kt < 5; ++kt){
    v16h a;
    int kb = kt * 32 + hi * 8;
#pragma unroll
    for (int i = 0; i < 16; ++i){
      int k = kb + (i & 7) + ((i & 8) ? 16 : 0);
      float v;
      if (k < 64)        v = srow[k];
      else if (k < 128)  v = trow[k - 64];
      else if (k == 128) v = nrm;
      else if (k < 145)  v = erow[k - 129];
      else               v = 0.0f;
      a[i] = (half_t)v;
    }
#pragma unroll
    for (int nt = 0; nt < 4; ++nt)
      acc1[nt] = wmma16(a, bfrag(We1f, lane, kt, nt), acc1[nt]);
  }

  silu_store_lds(acc1, ldsT[wid], lane);
  asm volatile("s_wait_dscnt 0" ::: "memory");
  v16h ah0 = afrag_lds(ldsT[wid], lane, 0);
  v16h ah1 = afrag_lds(ldsT[wid], lane, 1);

  // ---- GEMM2: message = silu(h @ W_e2 + b_e2)
  v8f accm[4];
#pragma unroll
  for (int nt = 0; nt < 4; ++nt){
    float b = be2[nt * 16 + n15];
#pragma unroll
    for (int r = 0; r < 8; ++r) accm[nt][r] = b;
  }
#pragma unroll
  for (int nt = 0; nt < 4; ++nt){
    accm[nt] = wmma16(ah0, bfrag(We2f, lane, 0, nt), accm[nt]);
    accm[nt] = wmma16(ah1, bfrag(We2f, lane, 1, nt), accm[nt]);
  }

  silu_store_lds(accm, ldsT[wid], lane);   // accm now holds message (f32)
  asm volatile("s_wait_dscnt 0" ::: "memory");
  v16h am0 = afrag_lds(ldsT[wid], lane, 0);
  v16h am1 = afrag_lds(ldsT[wid], lane, 1);

  // ---- GEMM3: c_h = silu(message @ W_c1 + b_c1)
  v8f accc[4];
#pragma unroll
  for (int nt = 0; nt < 4; ++nt){
    float b = bc1[nt * 16 + n15];
#pragma unroll
    for (int r = 0; r < 8; ++r) accc[nt][r] = b;
  }
#pragma unroll
  for (int nt = 0; nt < 4; ++nt){
    accc[nt] = wmma16(am0, bfrag(Wc1f, lane, 0, nt), accc[nt]);
    accc[nt] = wmma16(am1, bfrag(Wc1f, lane, 1, nt), accc[nt]);
  }

  // ---- 64->1 heads: coord_w = c_h @ W_c2 ; gate = sigmoid(message@W_i + b_i)
  float cw[8], g[8];
#pragma unroll
  for (int r = 0; r < 8; ++r){
    float pc = 0.0f, pg = 0.0f;
#pragma unroll
    for (int nt = 0; nt < 4; ++nt){
      pc += silu_f(accc[nt][r]) * Wc2[nt * 16 + n15];
      pg += accm[nt][r]         * Wi [nt * 16 + n15];
    }
    cw[r] = red16(pc);
    g[r]  = sigm_f(red16(pg) + bi[0]);
  }

  // ---- scatter: agg_message, coord_sum, cnt (f32 atomics)
#pragma unroll
  for (int r = 0; r < 8; ++r){
    int mm   = r + hi * 8;
    int sidx = startv[e0 + mm];
    float gr = g[r];
    size_t nb = (size_t)sidx * 64;
#pragma unroll
    for (int nt = 0; nt < 4; ++nt)
      atomicAdd(&agg[nb + nt * 16 + n15], accm[nt][r] * gr);
    if (n15 == 0){
      float cwv = cw[r];
      atomicAdd(&csum[(size_t)sidx * 3 + 0], ldsD[wid][mm * 4 + 0] * cwv);
      atomicAdd(&csum[(size_t)sidx * 3 + 1], ldsD[wid][mm * 4 + 1] * cwv);
      atomicAdd(&csum[(size_t)sidx * 3 + 2], ldsD[wid][mm * 4 + 2] * cwv);
      atomicAdd(&cnt[sidx], 1.0f);
    }
  }
}

// ---------------------------------------------------------------------------
// Node kernel: 1 wave = 16 nodes. coords/vels update + residual node MLP.
// ---------------------------------------------------------------------------
__global__ __launch_bounds__(256) void egnn_node_kernel(
    const float* __restrict__ nfi,
    const half_t* __restrict__ Wv1f, const float* __restrict__ bv1,
    const float* __restrict__ Wv2, const float* __restrict__ bv2,
    const half_t* __restrict__ Wn1f, const float* __restrict__ bn1,
    const half_t* __restrict__ Wn2f, const float* __restrict__ bn2,
    const float* __restrict__ agg, const float* __restrict__ csum,
    const float* __restrict__ cnt, float* __restrict__ out)
{
  __shared__ half_t ldsT[8][16 * 64];

  const int wid  = threadIdx.x >> 5;
  const int lane = threadIdx.x & 31;
  const int n0   = (blockIdx.x * 8 + wid) * 16;
  if (n0 >= NN) return;

  const int m    = lane & 15;
  const int hi   = (lane >> 4) & 1;
  const int n15  = lane & 15;
  const int node = n0 + m;
  const float* nrow = nfi + (size_t)node * ROW + NFOFF;
  const float* arow = agg + (size_t)node * 64;

  // ---- v_h = silu(nf @ W_v1 + b_v1); s = v_h @ W_v2 + b_v2
  v8f accv[4];
#pragma unroll
  for (int nt = 0; nt < 4; ++nt){
    float b = bv1[nt * 16 + n15];
#pragma unroll
    for (int r = 0; r < 8; ++r) accv[nt][r] = b;
  }
#pragma unroll
  for (int kt = 0; kt < 2; ++kt){
    v16h a;
    int kb = kt * 32 + hi * 8;
#pragma unroll
    for (int i = 0; i < 16; ++i){
      int k = kb + (i & 7) + ((i & 8) ? 16 : 0);
      a[i] = (half_t)nrow[k];
    }
#pragma unroll
    for (int nt = 0; nt < 4; ++nt)
      accv[nt] = wmma16(a, bfrag(Wv1f, lane, kt, nt), accv[nt]);
  }
  float sc[8];
#pragma unroll
  for (int r = 0; r < 8; ++r){
    float p = 0.0f;
#pragma unroll
    for (int nt = 0; nt < 4; ++nt)
      p += silu_f(accv[nt][r]) * Wv2[nt * 16 + n15];
    sc[r] = red16(p) + bv2[0];
  }

  // ---- coords/vels output (one lane per half-wave handles its 8 rows)
  if (n15 == 0){
#pragma unroll
    for (int r = 0; r < 8; ++r){
      int nd = n0 + r + hi * 8;
      const float* base = nfi + (size_t)nd * ROW;
      float c = fmaxf(cnt[nd], 1.0f);
#pragma unroll
      for (int j = 0; j < 3; ++j){
        float vj = base[3 + j];
        out[(size_t)nd * ROW + j]     = base[j] + csum[(size_t)nd * 3 + j] / c
                                        + sc[r] * vj;
        out[(size_t)nd * ROW + 3 + j] = vj;
      }
    }
  }

  // ---- nf update: nf += silu([nf,agg] @ W_n1 + b_n1) @ W_n2 + b_n2
  v8f accn[4];
#pragma unroll
  for (int nt = 0; nt < 4; ++nt){
    float b = bn1[nt * 16 + n15];
#pragma unroll
    for (int r = 0; r < 8; ++r) accn[nt][r] = b;
  }
#pragma unroll
  for (int kt = 0; kt < 4; ++kt){
    v16h a;
    int kb = kt * 32 + hi * 8;
#pragma unroll
    for (int i = 0; i < 16; ++i){
      int k = kb + (i & 7) + ((i & 8) ? 16 : 0);
      float v = (k < 64) ? nrow[k] : arow[k - 64];
      a[i] = (half_t)v;
    }
#pragma unroll
    for (int nt = 0; nt < 4; ++nt)
      accn[nt] = wmma16(a, bfrag(Wn1f, lane, kt, nt), accn[nt]);
  }
  silu_store_lds(accn, ldsT[wid], lane);
  asm volatile("s_wait_dscnt 0" ::: "memory");
  v16h an0 = afrag_lds(ldsT[wid], lane, 0);
  v16h an1 = afrag_lds(ldsT[wid], lane, 1);

  v8f acco[4];
#pragma unroll
  for (int nt = 0; nt < 4; ++nt){
    float b = bn2[nt * 16 + n15];
#pragma unroll
    for (int r = 0; r < 8; ++r) acco[nt][r] = b;
  }
#pragma unroll
  for (int nt = 0; nt < 4; ++nt){
    acco[nt] = wmma16(an0, bfrag(Wn2f, lane, 0, nt), acco[nt]);
    acco[nt] = wmma16(an1, bfrag(Wn2f, lane, 1, nt), acco[nt]);
  }
#pragma unroll
  for (int nt = 0; nt < 4; ++nt){
#pragma unroll
    for (int r = 0; r < 8; ++r){
      int nd  = n0 + r + hi * 8;
      int col = NFOFF + nt * 16 + n15;
      out[(size_t)nd * ROW + col] = nfi[(size_t)nd * ROW + col] + acco[nt][r];
    }
  }
}

// ---------------------------------------------------------------------------
extern "C" void kernel_launch(void* const* d_in, const int* in_sizes, int n_in,
                              void* d_out, int out_size, void* d_ws, size_t ws_size,
                              hipStream_t stream)
{
  (void)in_sizes; (void)n_in; (void)out_size; (void)ws_size;

  const float* nfi = (const float*)d_in[0];
  const int*   eidx= (const int*)  d_in[1];
  const float* ef  = (const float*)d_in[2];
  const float* We1 = (const float*)d_in[3];
  const float* be1 = (const float*)d_in[4];
  const float* We2 = (const float*)d_in[5];
  const float* be2 = (const float*)d_in[6];
  const float* Wc1 = (const float*)d_in[7];
  const float* bc1 = (const float*)d_in[8];
  const float* Wc2 = (const float*)d_in[9];
  const float* Wv1 = (const float*)d_in[10];
  const float* bv1 = (const float*)d_in[11];
  const float* Wv2 = (const float*)d_in[12];
  const float* bv2 = (const float*)d_in[13];
  const float* Wn1 = (const float*)d_in[14];
  const float* bn1 = (const float*)d_in[15];
  const float* Wn2 = (const float*)d_in[16];
  const float* bn2 = (const float*)d_in[17];
  const float* Wi  = (const float*)d_in[18];
  const float* bi  = (const float*)d_in[19];
  float* out = (float*)d_out;

  // Workspace layout: [agg NN*64][csum NN*3][cnt NN][f16 weight fragments]
  char* ws = (char*)d_ws;
  float*  agg  = (float*)ws;
  float*  csum = (float*)(ws + (size_t)NN * 64 * 4);
  float*  cnt  = (float*)(ws + (size_t)NN * 67 * 4);
  half_t* We1f = (half_t*)(ws + (size_t)NN * 68 * 4);  // 32B-aligned
  half_t* We2f = We1f + 5 * 4 * 32 * 16;
  half_t* Wc1f = We2f + 2 * 4 * 32 * 16;
  half_t* Wv1f = Wc1f + 2 * 4 * 32 * 16;
  half_t* Wn1f = Wv1f + 2 * 4 * 32 * 16;
  half_t* Wn2f = Wn1f + 4 * 4 * 32 * 16;

  // Zero the accumulation buffers (agg/csum/cnt are contiguous: NN*68 floats).
  {
    int nz = NN * 68;
    zero_f32_kernel<<<(nz + 255) / 256, 256, 0, stream>>>(agg, nz);
  }
  // Pack weights to f16 B-fragment order.
  pack_bfrag_kernel<<<(5 * 128 + 127) / 128, 128, 0, stream>>>(We1, We1f, 145, 5);
  pack_bfrag_kernel<<<(2 * 128 + 127) / 128, 128, 0, stream>>>(We2, We2f,  64, 2);
  pack_bfrag_kernel<<<(2 * 128 + 127) / 128, 128, 0, stream>>>(Wc1, Wc1f,  64, 2);
  pack_bfrag_kernel<<<(2 * 128 + 127) / 128, 128, 0, stream>>>(Wv1, Wv1f,  64, 2);
  pack_bfrag_kernel<<<(4 * 128 + 127) / 128, 128, 0, stream>>>(Wn1, Wn1f, 128, 4);
  pack_bfrag_kernel<<<(2 * 128 + 127) / 128, 128, 0, stream>>>(Wn2, Wn2f,  64, 2);

  // Edge phase: 1M edges, 16 edges/wave, 8 waves/block.
  {
    int waves  = NE / 16;              // 62500
    int blocks = (waves + 7) / 8;      // 7813
    egnn_edge_kernel<<<blocks, 256, 0, stream>>>(
        nfi, eidx, ef, We1f, be1, We2f, be2, Wc1f, bc1, Wc2, Wi, bi,
        agg, csum, cnt);
  }
  // Node phase: 50K nodes, 16 nodes/wave.
  {
    int waves  = (NN + 15) / 16;       // 3125
    int blocks = (waves + 7) / 8;      // 391
    egnn_node_kernel<<<blocks, 256, 0, stream>>>(
        nfi, Wv1f, bv1, Wv2, bv2, Wn1f, bn1, Wn2f, bn2,
        agg, csum, cnt, out);
  }
}
